// VolterraGraphConvLayer_26173530702105
// MI455X (gfx1250) — matrix operand order repaired and checked by reading, not compiled, not run on hardware
//
#include <hip/hip_runtime.h>

// ---------------------------------------------------------------------------
// Problem constants: N=8192 nodes, D=256 features, P=3 powers, ORDER=2
// ---------------------------------------------------------------------------
#define NN 8192
#define DD 256
#define PP 3
#define LHK (3 * DD)        // fused K of final GEMM: [X | S1 | S2]
#define KSPLIT 2
#define KHALF (NN / KSPLIT) // 4096

typedef __attribute__((ext_vector_type(16))) __bf16 v16bf;
typedef __attribute__((ext_vector_type(8)))  __bf16 v8bf;
typedef __attribute__((ext_vector_type(8)))  float  v8f;
typedef __attribute__((ext_vector_type(4)))  float  v4f;

// round-to-nearest-even f32 -> bf16
__device__ __forceinline__ __bf16 to_bf16(float f) {
    union { float f; unsigned u; } in; in.f = f;
    unsigned u = in.u + 0x7FFFu + ((in.u >> 16) & 1u);
    union { unsigned short s; __bf16 b; } out;
    out.s = (unsigned short)(u >> 16);
    return out.b;
}

__device__ __forceinline__ v8f wmma_bf16(v16bf a, v16bf b, v8f c) {
    return __builtin_amdgcn_wmma_f32_16x16x32_bf16(false, a, false, b, (short)0, c,
                                                   false, false);
}

// CDNA5 async global->LDS copy (16B per lane), tracked by ASYNCcnt.
// VDST = per-lane LDS byte offset, VADDR = per-lane 64-bit global address.
__device__ __forceinline__ void async_b128(unsigned lds_byte, const void* g) {
    asm volatile("global_load_async_to_lds_b128 %0, %1, off"
                 :: "v"(lds_byte), "v"((unsigned long long)(uintptr_t)g)
                 : "memory");
}
__device__ __forceinline__ void wait_asynccnt0() {
    asm volatile("s_wait_asynccnt 0x0" ::: "memory");
}

// ---------------------------------------------------------------------------
// Kernel 0: XbT[d][m] = bf16(X[m][d])  (LDS-tiled transpose, coalesced both ways)
// ---------------------------------------------------------------------------
__global__ void k_transpose_x(const float* __restrict__ X, __bf16* __restrict__ XbT) {
    __shared__ float tile[32][33];
    const int d0 = blockIdx.x * 32;
    const int m0 = blockIdx.y * 32;
    const int tx = threadIdx.x, ty = threadIdx.y;
#pragma unroll
    for (int i = ty; i < 32; i += 8)
        tile[i][tx] = X[(size_t)(m0 + i) * DD + d0 + tx];
    __syncthreads();
#pragma unroll
    for (int i = ty; i < 32; i += 8)
        XbT[(size_t)(d0 + i) * NN + m0 + tx] = to_bf16(tile[tx][i]);
}

// ---------------------------------------------------------------------------
// Kernel 1: Wb = bf16(W) (W[r][out][in] row-major IS the B^T layout we need),
//           bias[n] = b0 + 3*b1 + 6*b2 (combo counts for r=1,2)
// ---------------------------------------------------------------------------
__global__ void k_prep_w(const float* __restrict__ W, const float* __restrict__ b,
                         __bf16* __restrict__ Wb, float* __restrict__ bias) {
    const int idx = blockIdx.x * 256 + threadIdx.x;
    Wb[idx] = to_bf16(W[idx]);
    if (idx < DD)
        bias[idx] = b[idx] + 3.0f * b[DD + idx] + 6.0f * b[2 * DD + idx];
}

// ---------------------------------------------------------------------------
// Kernel 2: partial M_p = A_p[:, kz] @ X[kz, :]  (bf16 WMMA, fp32 acc)
// WG = 4 waves; wave owns 32 rows x 256 cols (acc = 256 VGPRs); WG = 128 rows.
// B tile (32x256 bf16, 16KB) staged in LDS in *fragment layout*, double
// buffered via GLOBAL_LOAD_ASYNC_TO_LDS (ASYNCcnt) so staging is neither
// ordered against the DS fragment reads nor holding VGPRs.
// ---------------------------------------------------------------------------
__global__ void __launch_bounds__(128)
k_gemm_ax(const float* __restrict__ A, const __bf16* __restrict__ XbT,
          float* __restrict__ Mpart) {
    // fragment layout: chunk (nt, lane) = 16 bf16 at element off (nt*32+lane)*16
    __shared__ __bf16 lds[2][16 * 32 * 16];   // 2 x 16 KB

    const int p    = blockIdx.y;
    const int kz   = blockIdx.z;
    const int tid  = threadIdx.x;
    const int wave = tid >> 5;
    const int lane = tid & 31;
    const int half = lane >> 4;
    const int l    = lane & 15;
    const int m0   = (blockIdx.x * 4 + wave) * 32;
    const int kbeg = kz * KHALF;

    const float* Ap = A + (size_t)p * NN * NN;
    float* Mp = Mpart + ((size_t)kz * PP + p) * NN * DD;

    v8f acc0[16] = {};   // rows m0 .. m0+15
    v8f acc1[16] = {};   // rows m0+16 .. m0+31

    const float* arow0 = Ap + (size_t)(m0 + l) * NN + kbeg;
    const float* arow1 = Ap + (size_t)(m0 + 16 + l) * NN + kbeg;

    // Cooperative B staging: 256 (nt,l) pairs, 2 per thread; each pair = 64B
    // of XbT row n = nt*16+l covering the 32-wide k window.
    const int pr0 = tid * 2;
    const int pr1 = tid * 2 + 1;
    const __bf16* gsrc0 = XbT + (size_t)((pr0 >> 4) * 16 + (pr0 & 15)) * NN + kbeg;
    const __bf16* gsrc1 = XbT + (size_t)((pr1 >> 4) * 16 + (pr1 & 15)) * NN + kbeg;
    // LDS byte offsets of the two 32B chunks each pair produces (k-half 0/1)
    const unsigned c0a = (unsigned)(((pr0 >> 4) * 32 + (pr0 & 15)) * 32);
    const unsigned c0b = (unsigned)(((pr0 >> 4) * 32 + (pr0 & 15) + 16) * 32);
    const unsigned c1a = (unsigned)(((pr1 >> 4) * 32 + (pr1 & 15)) * 32);
    const unsigned c1b = (unsigned)(((pr1 >> 4) * 32 + (pr1 & 15) + 16) * 32);
    const unsigned ldsbase[2] = { (unsigned)(uintptr_t)&lds[0][0],
                                  (unsigned)(uintptr_t)&lds[1][0] };

    // stage one 16KB B tile (k window at k0) into lds buffer `b`
    auto stage_tile = [&](int b, int k0) {
        const unsigned base = ldsbase[b];
        async_b128(base + c0a,      gsrc0 + k0);
        async_b128(base + c0a + 16, gsrc0 + k0 + 16);
        async_b128(base + c0b,      gsrc0 + k0 + 8);
        async_b128(base + c0b + 16, gsrc0 + k0 + 24);
        async_b128(base + c1a,      gsrc1 + k0);
        async_b128(base + c1a + 16, gsrc1 + k0 + 16);
        async_b128(base + c1b,      gsrc1 + k0 + 8);
        async_b128(base + c1b + 16, gsrc1 + k0 + 24);
    };

    // prologue: stage first tile
    stage_tile(0, 0);
    wait_asynccnt0();
    __syncthreads();

    for (int k0 = 0; k0 < KHALF; k0 += 32) {
        const int buf = (k0 >> 5) & 1;
        const bool more = (k0 + 32) < KHALF;

        // (1) kick off async staging of the NEXT tile into the other buffer
        //     (that buffer's previous readers retired at the last barrier)
        if (more) stage_tile(buf ^ 1, k0 + 32);

        // (2) A-fragment global loads + prefetch of the HBM stream
        const int kb = k0 + half * 8;
        v4f f0 = *(const v4f*)(arow0 + kb);
        v4f f1 = *(const v4f*)(arow0 + kb + 4);
        v4f f2 = *(const v4f*)(arow0 + kb + 16);
        v4f f3 = *(const v4f*)(arow0 + kb + 20);
        v4f f4 = *(const v4f*)(arow1 + kb);
        v4f f5 = *(const v4f*)(arow1 + kb + 4);
        v4f f6 = *(const v4f*)(arow1 + kb + 16);
        v4f f7 = *(const v4f*)(arow1 + kb + 20);
        __builtin_prefetch(arow0 + k0 + 512, 0, 0);   // global_prefetch_b8
        __builtin_prefetch(arow1 + k0 + 512, 0, 0);

        // (3) read the 16 B fragments of the current tile from LDS
        v16bf bfrag[16];
#pragma unroll
        for (int nt = 0; nt < 16; ++nt)
            bfrag[nt] = *(const v16bf*)(&lds[buf][(nt * 32 + lane) * 16]);

        // (4) convert A to bf16 fragments
        union { v16bf v; __bf16 e[16]; } ua0, ua1;
#pragma unroll
        for (int i = 0; i < 4; ++i) {
            ua0.e[i] = to_bf16(f0[i]); ua0.e[4 + i]  = to_bf16(f1[i]);
            ua0.e[8 + i] = to_bf16(f2[i]); ua0.e[12 + i] = to_bf16(f3[i]);
            ua1.e[i] = to_bf16(f4[i]); ua1.e[4 + i]  = to_bf16(f5[i]);
            ua1.e[8 + i] = to_bf16(f6[i]); ua1.e[12 + i] = to_bf16(f7[i]);
        }

        // (5) 32 back-to-back WMMAs (each staged B fragment feeds 2)
#pragma unroll
        for (int nt = 0; nt < 16; ++nt) {
            acc0[nt] = wmma_bf16(ua0.v, bfrag[nt], acc0[nt]);
            acc1[nt] = wmma_bf16(ua1.v, bfrag[nt], acc1[nt]);
        }

        // (6) our async writes must be in LDS before anyone passes the barrier
        wait_asynccnt0();
        __syncthreads();
    }

    // C/D layout: element v -> (m = base + half*8 + v, n = nt*16 + l)
#pragma unroll
    for (int nt = 0; nt < 16; ++nt) {
#pragma unroll
        for (int v = 0; v < 8; ++v) {
            const int n = nt * 16 + l;
            Mp[(size_t)(m0 + half * 8 + v) * DD + n]      = acc0[nt][v];
            Mp[(size_t)(m0 + 16 + half * 8 + v) * DD + n] = acc1[nt][v];
        }
    }
}

// ---------------------------------------------------------------------------
// Kernel 3: reduce K-split partials; LHS = [ X | S1 | S2 ] in bf16 (8192x768)
// S1 = M0+M1+M2 ; S2 = sum_{i<=j} Mi*Mj (6 Hadamard combos)
// ---------------------------------------------------------------------------
__global__ void k_build_lhs(const float* __restrict__ X, const float* __restrict__ M,
                            __bf16* __restrict__ LHSb) {
    const size_t idx = (size_t)blockIdx.x * 256 + threadIdx.x;   // over N*D
    const int m = (int)(idx / DD);
    const int d = (int)(idx % DD);
    const size_t plane = (size_t)NN * DD;
    const float x  = X[idx];
    const float a0 = M[idx]             + M[(size_t)PP * plane + idx];
    const float a1 = M[plane + idx]     + M[(size_t)PP * plane + plane + idx];
    const float a2 = M[2 * plane + idx] + M[(size_t)PP * plane + 2 * plane + idx];
    const float s1 = a0 + a1 + a2;
    const float s2 = a0 * a0 + a1 * a1 + a2 * a2 + a0 * a1 + a0 * a2 + a1 * a2;
    const size_t base = (size_t)m * LHK + d;
    LHSb[base]          = to_bf16(x);
    LHSb[base + DD]     = to_bf16(s1);
    LHSb[base + 2 * DD] = to_bf16(s2);
}

// ---------------------------------------------------------------------------
// Kernel 4: out = relu( LHSb @ [W0;W1;W2]^T + bias ), K = 768
// ---------------------------------------------------------------------------
__global__ void __launch_bounds__(128)
k_gemm_out(const __bf16* __restrict__ LHSb, const __bf16* __restrict__ Wb,
           const float* __restrict__ bias, float* __restrict__ out) {
    const int wave = threadIdx.x >> 5;
    const int lane = threadIdx.x & 31;
    const int half = lane >> 4;
    const int l    = lane & 15;
    const int m0   = (blockIdx.x * 4 + wave) * 16;

    v8f acc[16] = {};
    const __bf16* arow = LHSb + (size_t)(m0 + l) * LHK;

    for (int k0 = 0; k0 < LHK; k0 += 32) {
        const int kb = k0 + half * 16;
        union { v16bf v; v8bf h[2]; } ua;
        ua.h[0] = *(const v8bf*)(arow + kb);
        ua.h[1] = *(const v8bf*)(arow + kb + 8);

        const int kbb = k0 + half * 16;
        const int r   = kbb >> 8;         // W_r block (16 K stay in-block)
        const int dd0 = kbb & 255;
        v16bf bfrag[16];
#pragma unroll
        for (int nt = 0; nt < 16; ++nt)
            bfrag[nt] = *(const v16bf*)(Wb + (size_t)r * DD * DD
                                           + (size_t)(nt * 16 + l) * DD + dd0);
#pragma unroll
        for (int nt = 0; nt < 16; ++nt)
            acc[nt] = wmma_bf16(ua.v, bfrag[nt], acc[nt]);
    }

#pragma unroll
    for (int nt = 0; nt < 16; ++nt) {
#pragma unroll
        for (int v = 0; v < 8; ++v) {
            const int m = m0 + half * 8 + v;
            const int n = nt * 16 + l;
            const float r = acc[nt][v] + bias[n];
            out[(size_t)m * DD + n] = fmaxf(r, 0.0f);
        }
    }
}

// ---------------------------------------------------------------------------
// Workspace layout (bytes), ~64.5 MB total
// ---------------------------------------------------------------------------
#define OFF_XBT  ((size_t)0)                                    // 4 MiB
#define OFF_WB   ((size_t)(4 * 1024 * 1024))                    // 384 KiB
#define OFF_BIAS (OFF_WB + (size_t)3 * DD * DD * 2)             // 1 KiB
#define OFF_M    (OFF_BIAS + 1024)                              // 48 MiB (2 K-halves)
#define OFF_LHS  (OFF_M + (size_t)KSPLIT * PP * NN * DD * 4)    // 12 MiB

extern "C" void kernel_launch(void* const* d_in, const int* in_sizes, int n_in,
                              void* d_out, int out_size, void* d_ws, size_t ws_size,
                              hipStream_t stream) {
    const float* X = (const float*)d_in[0];   // [8192, 256]
    const float* A = (const float*)d_in[1];   // [3, 8192, 8192]
    const float* W = (const float*)d_in[2];   // [3, 256, 256]
    const float* b = (const float*)d_in[3];   // [3, 256]
    float* out = (float*)d_out;               // [8192, 256]

    char* ws = (char*)d_ws;
    __bf16* XbT  = (__bf16*)(ws + OFF_XBT);
    __bf16* Wb   = (__bf16*)(ws + OFF_WB);
    float*  bias = (float*)(ws + OFF_BIAS);
    float*  M    = (float*)(ws + OFF_M);
    __bf16* LHSb = (__bf16*)(ws + OFF_LHS);

    (void)in_sizes; (void)n_in; (void)out_size; (void)ws_size;

    k_transpose_x<<<dim3(DD / 32, NN / 32), dim3(32, 8), 0, stream>>>(X, XbT);
    k_prep_w<<<dim3((3 * DD * DD) / 256), dim3(256), 0, stream>>>(W, b, Wb, bias);
    k_gemm_ax<<<dim3(NN / 128, PP, KSPLIT), dim3(128), 0, stream>>>(A, XbT, M);
    k_build_lhs<<<dim3((NN * DD) / 256), dim3(256), 0, stream>>>(X, M, LHSb);
    k_gemm_out<<<dim3(NN / 64), dim3(128), 0, stream>>>(LHSb, Wb, bias, out);
}
